// PillarResNet18S_25958782337395
// MI455X (gfx1250) — matrix operand order repaired and checked by reading, not compiled, not run on hardware
//
#include <hip/hip_runtime.h>
#include <hip/hip_bf16.h>

// ---------------------------------------------------------------------------
// Types / helpers
// ---------------------------------------------------------------------------
typedef __attribute__((ext_vector_type(16))) __bf16    v16bf;
typedef __attribute__((ext_vector_type(8)))  float     v8f;
typedef __attribute__((ext_vector_type(8)))  unsigned  v8u;
typedef __attribute__((ext_vector_type(4)))  unsigned  u32x4;
typedef __attribute__((ext_vector_type(4)))  float     f32x4;

// 16-byte payload type + addrspace-qualified variants for async LDS DMA
typedef int b128_t __attribute__((vector_size(16)));
typedef __attribute__((address_space(1))) b128_t gb128; // global
typedef __attribute__((address_space(3))) b128_t sb128; // LDS

#if defined(__has_builtin)
#if __has_builtin(__builtin_amdgcn_global_load_async_to_lds_b128)
#define ASYNC_LDS 1
#endif
#endif
#ifndef ASYNC_LDS
#define ASYNC_LDS 0
#endif

// round-to-nearest-even f32 -> bf16
__device__ __forceinline__ unsigned short f2bf(float f) {
    unsigned u = __builtin_bit_cast(unsigned, f);
    u += 0x7FFFu + ((u >> 16) & 1u);
    return (unsigned short)(u >> 16);
}
__device__ __forceinline__ unsigned pack2(float lo, float hi) {
    return (unsigned)f2bf(lo) | ((unsigned)f2bf(hi) << 16);
}
__device__ __forceinline__ float bf2f(unsigned short h) {
    unsigned u = ((unsigned)h) << 16;
    return __builtin_bit_cast(float, u);
}
__device__ __forceinline__ v16bf frag_from(const unsigned* p0, const unsigned* p1) {
    u32x4 a = *(const u32x4*)p0;
    u32x4 b = *(const u32x4*)p1;
    v8u v;
    v[0] = a[0]; v[1] = a[1]; v[2] = a[2]; v[3] = a[3];
    v[4] = b[0]; v[5] = b[1]; v[6] = b[2]; v[7] = b[3];
    return __builtin_bit_cast(v16bf, v);
}
__device__ __forceinline__ void wait_async0() {
#if defined(__has_builtin) && __has_builtin(__builtin_amdgcn_s_wait_asynccnt)
    __builtin_amdgcn_s_wait_asynccnt(0);
#else
    asm volatile("s_wait_asynccnt 0" ::: "memory");
#endif
}

// ---------------------------------------------------------------------------
// Prep kernels
// ---------------------------------------------------------------------------
__global__ void fold_bn_kernel(const float* __restrict__ gamma,
                               const float* __restrict__ beta,
                               const float* __restrict__ mean,
                               const float* __restrict__ var,
                               float* __restrict__ scale,
                               float* __restrict__ bias, int C) {
    int c = blockIdx.x * blockDim.x + threadIdx.x;
    if (c < C) {
        float s = gamma[c] * rsqrtf(var[c] + 1e-3f);
        scale[c] = s;
        bias[c]  = beta[c] - mean[c] * s;
    }
}

// HWIO f32 weights -> per-lane WMMA bf16 B-fragment layout:
// wfrag[((tap*KC + kc)*NC + nc)*32 + lane][8 dwords]; lane = half*16 + n16;
// dword j holds k = kc*32 + 2j(+1) + half*16  (contiguous k order, same as the A fragment)
__global__ void wtrans_kernel(const float* __restrict__ w,
                              unsigned* __restrict__ wfrag, int CIN, int COUT) {
    int KC = CIN / 32, NC = COUT / 16;
    int idx = blockIdx.x * blockDim.x + threadIdx.x;
    int total = 9 * KC * NC * 32;
    if (idx >= total) return;
    int lane = idx & 31;
    int rest = idx >> 5;
    int nc = rest % NC; rest /= NC;
    int kc = rest % KC;
    int tap = rest / KC;
    int n = (lane & 15) + nc * 16;
    int half = lane >> 4;
    unsigned* dst = wfrag + (size_t)idx * 8;
#pragma unroll
    for (int j = 0; j < 8; ++j) {
        int kb = kc * 32 + 2 * j + half * 16;
        float lo = w[((size_t)tap * CIN + kb)     * COUT + n];
        float hi = w[((size_t)tap * CIN + kb + 1) * COUT + n];
        dst[j] = pack2(lo, hi);
    }
}

// x*m -> bf16 activation buffer; mask -> float m1
__global__ void maskx_kernel(const float* __restrict__ x, const int* __restrict__ mask,
                             unsigned short* __restrict__ xa, float* __restrict__ m1,
                             int npix) {
    int p = blockIdx.x * blockDim.x + threadIdx.x;
    if (p >= npix) return;
    float mv = mask[p] ? 1.0f : 0.0f;
    m1[p] = mv;
    const f32x4* src = (const f32x4*)(x + (size_t)p * 32);
    unsigned* dst = (unsigned*)(xa + (size_t)p * 32);
#pragma unroll
    for (int i = 0; i < 8; ++i) {
        f32x4 v = src[i];
        v *= mv;
        dst[2 * i + 0] = pack2(v[0], v[1]);
        dst[2 * i + 1] = pack2(v[2], v[3]);
    }
}

// 3x3 / stride-2 / pad-1 max-pool of the mask (active-set dilation)
__global__ void pool_kernel(const float* __restrict__ mi, float* __restrict__ mo,
                            int B, int Ho, int Wo) {
    int idx = blockIdx.x * blockDim.x + threadIdx.x;
    int total = B * Ho * Wo;
    if (idx >= total) return;
    int ox = idx % Wo;
    int r = idx / Wo;
    int oy = r % Ho;
    int b  = r / Ho;
    int Hi = Ho * 2, Wi = Wo * 2;
    float v = 0.0f;
#pragma unroll
    for (int ky = 0; ky < 3; ++ky) {
        int iy = oy * 2 + ky - 1;
        if ((unsigned)iy >= (unsigned)Hi) continue;
#pragma unroll
        for (int kx = 0; kx < 3; ++kx) {
            int ix = ox * 2 + kx - 1;
            if ((unsigned)ix >= (unsigned)Wi) continue;
            v = fmaxf(v, mi[((size_t)b * Hi + iy) * Wi + ix]);
        }
    }
    mo[idx] = v;
}

// ---------------------------------------------------------------------------
// Implicit-GEMM 3x3 conv, bf16 WMMA, fused BN/mask/residual/ReLU epilogue.
// One LDS staging of 3 input rows per workgroup; all 9 taps consume it.
// ---------------------------------------------------------------------------
template <int CIN, int COUT, int STRIDE, int HW>
__global__ void __launch_bounds__(256, 2)
conv_wmma(const unsigned short* __restrict__ in,
          const unsigned* __restrict__ wfrag,
          const float* __restrict__ scale, const float* __restrict__ bias,
          const float* __restrict__ mk, const unsigned short* __restrict__ res,
          unsigned short* __restrict__ outb, float* __restrict__ outf) {
    constexpr int NW_N   = COUT / 32;           // waves along N
    constexpr int NW_M   = 8 / NW_N;            // waves along M
    constexpr int TILE_M = 16 * NW_M;           // output pixels per workgroup
    constexpr int KC     = CIN / 32;            // K chunks per tap
    constexpr int NC     = COUT / 16;           // B fragments along N
    constexpr int WPIX   = STRIDE * TILE_M + 2; // staged pixels per row
    constexpr int CP     = CIN + 8;             // padded channel stride (bank-conflict-free)
    constexpr int Hin    = HW * STRIDE, Win = HW * STRIDE;
    constexpr int PC8    = CIN / 8;             // 8-elem chunks per pixel
    constexpr int E8     = 3 * WPIX * PC8;      // total staged chunks
    constexpr int TPR    = HW / TILE_M;         // tiles per output row

    __shared__ unsigned short As[3 * WPIX * CP]; // bf16, [row][pixel][channel]

    const int tid   = threadIdx.x;
    const int lane  = tid & 31;
    const int wave  = tid >> 5;
    const int waveM = wave % NW_M;
    const int waveN = wave / NW_M;

    const int x0  = (blockIdx.x % TPR) * TILE_M;
    const int row = blockIdx.x / TPR;
    const int oy  = row % HW;
    const int b   = row / HW;

    // ---- stage 3 input rows (with zero padding) into LDS -------------------
    for (int i = tid; i < E8; i += 256) {
        const int c    = (i % PC8) * 8;
        const int rest = i / PC8;
        const int px   = rest % WPIX;
        const int r    = rest / WPIX;
        const int iy   = oy * STRIDE + r - 1;
        const int ix   = x0 * STRIDE + px - 1;
        unsigned short* dst = &As[(r * WPIX + px) * CP + c];
        const bool valid =
            (unsigned)iy < (unsigned)Hin && (unsigned)ix < (unsigned)Win;
#if ASYNC_LDS
        if (valid) {
            const unsigned short* src =
                in + (((size_t)b * Hin + iy) * Win + ix) * (size_t)CIN + c;
            __builtin_amdgcn_global_load_async_to_lds_b128(
                (gb128*)src, (sb128*)dst, 0, 0);
        } else {
            u32x4 z = {};
            *(u32x4*)dst = z;
        }
#else
        u32x4 v = {};
        if (valid)
            v = *(const u32x4*)(in +
                                (((size_t)b * Hin + iy) * Win + ix) * (size_t)CIN + c);
        *(u32x4*)dst = v;
#endif
    }
#if ASYNC_LDS
    wait_async0();
#endif
    __syncthreads();

    // ---- 9 taps x KC chunks of WMMA ---------------------------------------
    v8f acc0 = {};
    v8f acc1 = {};
    const int m15  = lane & 15;
    const int half = lane >> 4;
    const int pixBase = STRIDE * (waveM * 16 + m15) + 1;

#pragma unroll
    for (int tap = 0; tap < 9; ++tap) {
        const int dy = tap / 3 - 1;
        const int dx = tap % 3 - 1;
#pragma unroll
        for (int kc = 0; kc < KC; ++kc) {
            // A fragment: dwords 0-7 = k[kb..kb+15],
            // kb = kc*32 + half*16
            const unsigned* ap =
                (const unsigned*)As +
                ((((dy + 1) * WPIX + pixBase + dx) * CP + kc * 32 + half * 16) >> 1);
            v16bf Af = frag_from(ap, ap + 4);
#pragma unroll
            for (int s = 0; s < 2; ++s) {
                const unsigned* bp =
                    wfrag +
                    ((((size_t)tap * KC + kc) * NC + waveN * 2 + s) * 32 + lane) * 8;
                v16bf Bf = frag_from(bp, bp + 4);
                if (s == 0)
                    acc0 = __builtin_amdgcn_wmma_f32_16x16x32_bf16(
                        false, Af, false, Bf, (short)0, acc0, false, false);
                else
                    acc1 = __builtin_amdgcn_wmma_f32_16x16x32_bf16(
                        false, Af, false, Bf, (short)0, acc1, false, false);
            }
        }
    }

    // ---- epilogue: scale/bias, mask, residual, ReLU -----------------------
    // C/D layout: VGPR r -> lanes 0-15 (M=r, N=lane), lanes 16-31 (M=r+8)
    const size_t rowBase = ((size_t)b * HW + oy) * HW + x0 + waveM * 16;
    float mval[8];
#pragma unroll
    for (int r = 0; r < 8; ++r) mval[r] = mk[rowBase + r + half * 8];

#pragma unroll
    for (int s = 0; s < 2; ++s) {
        const int ch = waveN * 32 + s * 16 + m15;
        const float scv = scale[ch];
        const float biv = bias[ch];
        const v8f acc = (s == 0) ? acc0 : acc1;
#pragma unroll
        for (int r = 0; r < 8; ++r) {
            const size_t p = rowBase + r + half * 8;
            float v = acc[r] * scv + biv;
            v *= mval[r];
            if (res) v += bf2f(res[p * COUT + ch]); // same-index RAW: race-free
            v = fmaxf(v, 0.0f);
            if (outb) outb[p * COUT + ch] = f2bf(v);
            if (outf) outf[p * COUT + ch] = v;
        }
    }
}

// ---------------------------------------------------------------------------
// Host side
// ---------------------------------------------------------------------------
static void launch_conv(int cin, int cout, int stride, int HW,
                        const unsigned short* in, const unsigned* wf,
                        const float* sc, const float* bi, const float* mk,
                        const unsigned short* res, unsigned short* outb,
                        float* outf, hipStream_t stream) {
    int tileM = 16 * (8 / (cout / 32));
    dim3 grid(4 * HW * (HW / tileM));
#define CASE(CI, CO, ST, HH)                                                  \
    if (cin == CI && cout == CO && stride == ST && HW == HH) {                \
        conv_wmma<CI, CO, ST, HH>                                             \
            <<<grid, 256, 0, stream>>>(in, wf, sc, bi, mk, res, outb, outf);  \
        return;                                                               \
    }
    CASE(32, 32, 1, 512)
    CASE(32, 64, 2, 256)
    CASE(64, 64, 1, 256)
    CASE(64, 128, 2, 128)
    CASE(128, 128, 1, 128)
    CASE(128, 256, 2, 64)
    CASE(256, 256, 1, 64)
#undef CASE
}

extern "C" void kernel_launch(void* const* d_in, const int* in_sizes, int n_in,
                              void* d_out, int out_size, void* d_ws, size_t ws_size,
                              hipStream_t stream) {
    (void)in_sizes; (void)n_in; (void)out_size; (void)ws_size;

    const float* x    = (const float*)d_in[0];
    const int*   mask = (const int*)d_in[1];

    // d_out = concat(x1, x2, x3, x4) f32
    float* F1 = (float*)d_out;        // 4*512*512*32
    float* F2 = F1 + 33554432;        // 4*256*256*64
    float* F3 = F2 + 16777216;        // 4*128*128*128
    float* F4 = F3 + 8388608;         // 4*64*64*256

    // workspace: bf16 ping-pong activations + masks + folded BN + weight frags
    unsigned short* Abuf = (unsigned short*)d_ws;   // 33554432 bf16
    unsigned short* Bbuf = Abuf + 33554432;         // 33554432 bf16
    float* m1 = (float*)(Bbuf + 33554432);          // 1048576
    float* m2 = m1 + 1048576;                       // 262144
    float* m3 = m2 + 262144;                        // 65536
    float* m4 = m3 + 65536;                         // 16384
    float* scalesAll = m4 + 16384;                  // 20*256
    float* biasesAll = scalesAll + 20 * 256;        // 20*256
    unsigned* wfragBase = (unsigned*)(biasesAll + 20 * 256);

    // conv_bn groups, source insertion order:
    // 0-4: s1.bv.c0,c1,c2, s1.b1.c1,c2 ; 5-9: s2.down,b1.c1,b1.c2,b2.c1,b2.c2
    // 10-14: s3 ; 15-19: s4
    const int gin[20]  = {32, 32, 32, 32, 32, 32, 64, 64, 64, 64,
                          64, 128, 128, 128, 128, 128, 256, 256, 256, 256};
    const int gout[20] = {32, 32, 32, 32, 32, 64, 64, 64, 64, 64,
                          128, 128, 128, 128, 128, 256, 256, 256, 256, 256};

    unsigned* wfrag_g[20];
    size_t woff = 0;
    for (int g = 0; g < 20; ++g) {
        const float* w   = (const float*)d_in[2 + 5 * g + 0];
        const float* gam = (const float*)d_in[2 + 5 * g + 1];
        const float* bet = (const float*)d_in[2 + 5 * g + 2];
        const float* mea = (const float*)d_in[2 + 5 * g + 3];
        const float* var = (const float*)d_in[2 + 5 * g + 4];
        fold_bn_kernel<<<1, 256, 0, stream>>>(gam, bet, mea, var,
                                              scalesAll + g * 256,
                                              biasesAll + g * 256, gout[g]);
        wfrag_g[g] = wfragBase + woff;
        int total = 9 * (gin[g] / 32) * (gout[g] / 16) * 32;
        wtrans_kernel<<<(total + 255) / 256, 256, 0, stream>>>(w, wfrag_g[g],
                                                               gin[g], gout[g]);
        woff += (size_t)9 * gin[g] * gout[g] / 2;
    }

    // masked input (bf16) + mask pyramid
    maskx_kernel<<<(1048576 + 255) / 256, 256, 0, stream>>>(x, mask, Abuf, m1, 1048576);
    pool_kernel<<<(4 * 256 * 256 + 255) / 256, 256, 0, stream>>>(m1, m2, 4, 256, 256);
    pool_kernel<<<(4 * 128 * 128 + 255) / 256, 256, 0, stream>>>(m2, m3, 4, 128, 128);
    pool_kernel<<<(4 * 64 * 64 + 255) / 256, 256, 0, stream>>>(m3, m4, 4, 64, 64);

    struct LL {
        int g, cin, cout, stride, HW;
        const unsigned short *in, *res;
        const float* mk;
        unsigned short* outb;
        float* outf;
    };
    const LL Ls[20] = {
        // stage1 @512, C=32
        {0, 32, 32, 1, 512, Abuf, nullptr, m1, Bbuf, nullptr},   // bv.c0
        {1, 32, 32, 1, 512, Bbuf, nullptr, m1, Abuf, nullptr},   // bv.c1
        {2, 32, 32, 1, 512, Abuf, Bbuf,    m1, Bbuf, nullptr},   // bv.c2 (+res)
        {3, 32, 32, 1, 512, Bbuf, nullptr, m1, Abuf, nullptr},   // b1.c1
        {4, 32, 32, 1, 512, Abuf, Bbuf,    m1, Bbuf, F1},        // b1.c2 -> feat1
        // stage2 @256, C=64
        {5, 32, 64, 2, 256, Bbuf, nullptr, m2, Abuf, nullptr},   // down
        {6, 64, 64, 1, 256, Abuf, nullptr, m2, Bbuf, nullptr},
        {7, 64, 64, 1, 256, Bbuf, Abuf,    m2, Abuf, nullptr},
        {8, 64, 64, 1, 256, Abuf, nullptr, m2, Bbuf, nullptr},
        {9, 64, 64, 1, 256, Bbuf, Abuf,    m2, Abuf, F2},        // -> feat2
        // stage3 @128, C=128
        {10, 64, 128, 2, 128, Abuf, nullptr, m3, Bbuf, nullptr},
        {11, 128, 128, 1, 128, Bbuf, nullptr, m3, Abuf, nullptr},
        {12, 128, 128, 1, 128, Abuf, Bbuf,    m3, Bbuf, nullptr},
        {13, 128, 128, 1, 128, Bbuf, nullptr, m3, Abuf, nullptr},
        {14, 128, 128, 1, 128, Abuf, Bbuf,    m3, Bbuf, F3},     // -> feat3
        // stage4 @64, C=256
        {15, 128, 256, 2, 64, Bbuf, nullptr, m4, Abuf, nullptr},
        {16, 256, 256, 1, 64, Abuf, nullptr, m4, Bbuf, nullptr},
        {17, 256, 256, 1, 64, Bbuf, Abuf,    m4, Abuf, nullptr},
        {18, 256, 256, 1, 64, Abuf, nullptr, m4, Bbuf, nullptr},
        {19, 256, 256, 1, 64, Bbuf, Abuf,    m4, Abuf, F4},      // -> feat4
    };

    for (int i = 0; i < 20; ++i) {
        const LL& L = Ls[i];
        launch_conv(L.cin, L.cout, L.stride, L.HW, L.in, wfrag_g[L.g],
                    scalesAll + L.g * 256, biasesAll + L.g * 256, L.mk, L.res,
                    L.outb, L.outf, stream);
    }
}
